// MultiHeadAttention_11098195492975
// MI455X (gfx1250) — matrix operand order repaired and checked by reading, not compiled, not run on hardware
//
#include <hip/hip_runtime.h>
#include <hip/hip_bf16.h>
#include <math.h>

// MHA forward for MI455X (gfx1250, wave32, WMMA).
// out = softmax(Q K^T / 32) @ V  +  mask @ V   (mask added post-softmax)
// Matrix math via v_wmma_f32_16x16x32_f16; K/V staged through LDS with
// async global->LDS copies (ASYNCcnt) when the toolchain exposes them.

#define D_MODEL 1024
#define N_HEAD  16
#define HEAD_DIM 64
#define BATCH   4
#define SEQ     1024

typedef __attribute__((ext_vector_type(16))) _Float16 v16h;
typedef __attribute__((ext_vector_type(8)))  _Float16 v8h;
typedef __attribute__((ext_vector_type(8)))  float    v8f;
typedef __attribute__((ext_vector_type(4)))  float    v4f;
typedef __attribute__((ext_vector_type(4)))  int      int4v;
typedef __attribute__((ext_vector_type(4)))  unsigned int uint4v;

#define WMMA_F16(a, b, c) \
  __builtin_amdgcn_wmma_f32_16x16x32_f16(false, (a), false, (b), (short)0, (c), false, false)

// ---- async global->LDS copy (16B per lane), guarded for portability -------
#if __has_builtin(__builtin_amdgcn_global_load_async_to_lds_b128)
#define HAVE_ASYNC_LDS 1
#else
#define HAVE_ASYNC_LDS 0
#endif

typedef __attribute__((address_space(1))) int4v gint4;   // global int4
typedef __attribute__((address_space(3))) int4v lint4;   // LDS int4

static __device__ __forceinline__ void cp16_to_lds(const void* g, void* l) {
#if HAVE_ASYNC_LDS
  __builtin_amdgcn_global_load_async_to_lds_b128((gint4*)g, (lint4*)l, 0, 0);
#else
  *(uint4v*)l = *(const uint4v*)g;
#endif
}

static __device__ __forceinline__ void wait_cp_lds() {
#if HAVE_ASYNC_LDS
#if __has_builtin(__builtin_amdgcn_s_wait_asynccnt)
  __builtin_amdgcn_s_wait_asynccnt(0);
#else
  asm volatile("s_wait_asynccnt 0x0" ::: "memory");
#endif
#endif
}

// A-fragment (16x32, M x K), A row-major with leading dim ld.
// Lane L holds row M = L&15, K in {kb..kb+15}, kb = (L>>4)*16.
static __device__ __forceinline__ v16h load_a_frag(const _Float16* __restrict__ base,
                                                   int ld, int lane) {
  const int m  = lane & 15;
  const int kb = (lane >> 4) << 4;
  v8h lo = *(const v8h*)(base + (size_t)m * ld + kb);
  v8h hi = *(const v8h*)(base + (size_t)m * ld + kb + 8);
  return __builtin_shufflevector(lo, hi, 0,1,2,3,4,5,6,7,8,9,10,11,12,13,14,15);
}

// B-fragment (32x16, K x N) where B[k][n] = base[n*ld + k] (column-n contiguous
// over k, i.e. W row-major for x @ W^T). Lane L: n = L&15, K window (L>>4)*16.
static __device__ __forceinline__ v16h load_b_frag(const _Float16* __restrict__ base,
                                                   int ld, int lane) {
  const int n  = lane & 15;
  const int kb = (lane >> 4) << 4;
  v8h lo = *(const v8h*)(base + (size_t)n * ld + kb);
  v8h hi = *(const v8h*)(base + (size_t)n * ld + kb + 8);
  return __builtin_shufflevector(lo, hi, 0,1,2,3,4,5,6,7,8,9,10,11,12,13,14,15);
}

// ---------------------------------------------------------------- convert
__global__ void cvt_f32_to_f16(const float* __restrict__ in,
                               _Float16* __restrict__ out, int n) {
  int i = blockIdx.x * blockDim.x + threadIdx.x;
  const int stride = gridDim.x * blockDim.x;
  for (; i < n; i += stride) out[i] = (_Float16)in[i];
}

// ------------------------------------------------------- projection GEMM
// 32x64 output tile per wave: 2 A-frags x 4 B-frags -> 8 WMMAs per K-step
// (2.7x better L2 arithmetic intensity than 16x16/wave).
// out[n,o] = sum_d act[n,d] * W[o,d] + bias[o], head-major f16 output:
// store_transposed==0 -> [b,h,s,hd] (Q,K); ==1 -> [b,h,hd,s] (V).
__global__ __launch_bounds__(128) void proj_gemm_f16(
    const _Float16* __restrict__ actH,   // [4096, 1024]
    const _Float16* __restrict__ wH,     // [1024, 1024] row o, contiguous over d
    const float*    __restrict__ bias,   // [1024] f32
    _Float16* __restrict__ outH,
    int store_transposed)
{
  const int lane = threadIdx.x & 31;
  const int wid  = blockIdx.x * (blockDim.x >> 5) + (threadIdx.x >> 5);
  const int ot = wid & 15;      // 16 col tiles of 64
  const int nt = wid >> 4;      // 128 row tiles of 32
  const _Float16* Abase = actH + (size_t)nt * 32 * D_MODEL;
  const _Float16* Bbase = wH   + (size_t)ot * 64 * D_MODEL;

  v8f c[2][4] = {};
  for (int kt = 0; kt < D_MODEL / 32; ++kt) {
    __builtin_prefetch(Abase + (kt + 2) * 32, 0, 1);
    __builtin_prefetch(Bbase + (kt + 2) * 32, 0, 1);
    v16h a0 = load_a_frag(Abase + kt * 32,                D_MODEL, lane);
    v16h a1 = load_a_frag(Abase + 16 * D_MODEL + kt * 32, D_MODEL, lane);
    #pragma unroll
    for (int j = 0; j < 4; ++j) {
      v16h bj = load_b_frag(Bbase + (size_t)j * 16 * D_MODEL + kt * 32, D_MODEL, lane);
      c[0][j] = WMMA_F16(a0, bj, c[0][j]);
      c[1][j] = WMMA_F16(a1, bj, c[1][j]);
    }
  }

  const int half = lane >> 4;
  #pragma unroll
  for (int j = 0; j < 4; ++j) {
    const int o  = ot * 64 + j * 16 + (lane & 15);
    const float bv = bias[o];
    const int h  = o >> 6;
    const int hd = o & 63;
    #pragma unroll
    for (int mi = 0; mi < 2; ++mi) {
      if (!store_transposed) {
        #pragma unroll
        for (int r = 0; r < 8; ++r) {
          const int n  = nt * 32 + mi * 16 + r + 8 * half;
          const int bb = n >> 10, s = n & 1023;
          outH[(((size_t)bb * N_HEAD + h) * SEQ + s) * HEAD_DIM + hd] =
              (_Float16)(c[mi][j][r] + bv);
        }
      } else {
        const int n0 = nt * 32 + mi * 16 + 8 * half;
        const int bb = n0 >> 10, s0 = n0 & 1023;
        v8h pack;
        #pragma unroll
        for (int r = 0; r < 8; ++r) pack[r] = (_Float16)(c[mi][j][r] + bv);
        *(v8h*)(outH + (((size_t)bb * N_HEAD + h) * HEAD_DIM + hd) * SEQ + s0) = pack;
      }
    }
  }
}

// --------------------------------------------------------- flash attention
// Block = 4 waves = one (b,h) with 4 consecutive q-tiles. Each iteration the
// block cooperatively stages the 32-key K-tile (32x64 f16) and V-tile (64x32
// f16, from transposed V) into LDS via async global->LDS copies, then every
// wave computes L^T = K*Q^T (keys in M, queries in N) so the softmax'd
// C-fragment maps onto the B-fragment of P^T with one half-swap shuffle,
// and O^T = V^T * P^T accumulates in f32.
__global__ __launch_bounds__(128) void flash_attn_f16(
    const _Float16* __restrict__ Qh,  // [B,H,S,64]
    const _Float16* __restrict__ Kh,  // [B,H,S,64]
    const _Float16* __restrict__ Vt,  // [B,H,64,S]
    float* __restrict__ out)          // [B,S,1024]
{
  __shared__ __align__(16) _Float16 sK[32 * 64];   // keys x hd
  __shared__ __align__(16) _Float16 sV[64 * 32];   // hd x keys

  const int tid  = threadIdx.x;
  const int lane = tid & 31;
  const int wave = tid >> 5;
  const int bid  = blockIdx.x;             // 1024 = B*H*16
  const int qg = bid & 15;                 // q-group of 4 tiles
  const int h  = (bid >> 4) & (N_HEAD - 1);
  const int b  = bid >> 8;
  const int qt = qg * 4 + wave;
  const size_t bh = (size_t)b * N_HEAD + h;
  const _Float16* Qbase = Qh + (bh * SEQ + (size_t)qt * 16) * HEAD_DIM;
  const _Float16* Kbase = Kh + bh * SEQ * HEAD_DIM;
  const _Float16* Vbase = Vt + bh * HEAD_DIM * SEQ;

  // Q^T B-fragments (hd 0..31, 32..63) — loop invariant.
  const v16h bQ0 = load_b_frag(Qbase,      HEAD_DIM, lane);
  const v16h bQ1 = load_b_frag(Qbase + 32, HEAD_DIM, lane);

  v8f O0 = {}, O1 = {}, O2 = {}, O3 = {};
  float mrun = -__builtin_inff(), lrun = 0.f;
  const float scale = 0.03125f;                 // 1/sqrt(d_model)
  const float LOG2E = 1.4426950408889634f;
  const bool lowhalf = (lane < 16);

  for (int kt = 0; kt < SEQ / 32; ++kt) {
    __syncthreads();   // previous iteration's consumers done
    // Stage K-tile: 4 KB contiguous (rows kt*32..+31 of this head).
    {
      const _Float16* gK = Kbase + (size_t)(kt * 32) * HEAD_DIM;
      #pragma unroll
      for (int c = 0; c < 2; ++c) {
        const int chunk = tid + c * 128;       // 256 chunks of 16B
        cp16_to_lds(gK + chunk * 8, sK + chunk * 8);
      }
      // Stage V-tile: 64 rows (hd) x 32 keys from strided Vt rows.
      #pragma unroll
      for (int c = 0; c < 2; ++c) {
        const int chunk = tid + c * 128;       // 256 chunks of 16B
        const int row = chunk >> 2, off = (chunk & 3) * 8;
        cp16_to_lds(Vbase + (size_t)row * SEQ + kt * 32 + off,
                    sV + row * 32 + off);
      }
    }
    wait_cp_lds();
    __syncthreads();   // staged K/V visible to all 4 waves

    // logits^T, 32 keys (two 16-key M tiles) x 16 queries, from LDS
    v8f z = {};
    v8f lt0 = WMMA_F16(load_a_frag(sK,                HEAD_DIM, lane), bQ0, z);
    lt0     = WMMA_F16(load_a_frag(sK + 32,           HEAD_DIM, lane), bQ1, lt0);
    v8f lt1 = WMMA_F16(load_a_frag(sK + 16 * 64,      HEAD_DIM, lane), bQ0, z);
    lt1     = WMMA_F16(load_a_frag(sK + 16 * 64 + 32, HEAD_DIM, lane), bQ1, lt1);

    // online softmax over keys (M dim): 16 vgpr slots + half-swap shuffle
    float tmax = -__builtin_inff();
    #pragma unroll
    for (int r = 0; r < 8; ++r) {
      lt0[r] *= scale; lt1[r] *= scale;
      tmax = fmaxf(tmax, fmaxf(lt0[r], lt1[r]));
    }
    tmax = fmaxf(tmax, __shfl_xor(tmax, 16, 32));
    const float mnew  = fmaxf(mrun, tmax);
    const float alpha = exp2f((mrun - mnew) * LOG2E);
    float rsum = 0.f;
    #pragma unroll
    for (int r = 0; r < 8; ++r) {
      lt0[r] = exp2f((lt0[r] - mnew) * LOG2E);
      lt1[r] = exp2f((lt1[r] - mnew) * LOG2E);
      rsum += lt0[r] + lt1[r];
    }
    rsum += __shfl_xor(rsum, 16, 32);
    lrun = lrun * alpha + rsum;
    mrun = mnew;
    #pragma unroll
    for (int r = 0; r < 8; ++r) { O0[r] *= alpha; O1[r] *= alpha; O2[r] *= alpha; O3[r] *= alpha; }

    // P^T (32 keys x 16 q) as a B-fragment: C-layout halves hold keys
    // {0..7,16..23} (low) / {8..15,24..31} (high); shfl_xor(16) completes it.
    v16h bP;
    #pragma unroll
    for (int j = 0; j < 8; ++j) {
      const float s0 = __shfl_xor(lt0[j], 16, 32);
      const float s1 = __shfl_xor(lt1[j], 16, 32);
      bP[j]     = (_Float16)(lowhalf ? lt0[j] : s1);
      bP[j + 8] = (_Float16)(lowhalf ? s0 : lt1[j]);
    }

    // O^T(64 x 16q) += V^T(64 x 32k) @ P^T(32k x 16q), V^T from LDS
    O0 = WMMA_F16(load_a_frag(sV,           32, lane), bP, O0);
    O1 = WMMA_F16(load_a_frag(sV + 16 * 32, 32, lane), bP, O1);
    O2 = WMMA_F16(load_a_frag(sV + 32 * 32, 32, lane), bP, O2);
    O3 = WMMA_F16(load_a_frag(sV + 48 * 32, 32, lane), bP, O3);
  }

  // normalize and store: O^T element (hd = t*16 + 8*half + r, q = lane&15)
  const float invl = 1.f / lrun;
  const int q = lane & 15;
  const int half8 = (lane >> 4) * 8;
  float* orow = out + ((size_t)b * SEQ + (size_t)qt * 16 + q) * D_MODEL
                    + h * HEAD_DIM + half8;
  #define STORE_O(Ot, t)                                                      \
    { v4f p0 = { Ot[0]*invl, Ot[1]*invl, Ot[2]*invl, Ot[3]*invl };            \
      v4f p1 = { Ot[4]*invl, Ot[5]*invl, Ot[6]*invl, Ot[7]*invl };            \
      *(v4f*)(orow + (t)*16 + 0) = p0;                                        \
      *(v4f*)(orow + (t)*16 + 4) = p1; }
  STORE_O(O0, 0) STORE_O(O1, 1) STORE_O(O2, 2) STORE_O(O3, 3)
  #undef STORE_O
}

// ------------------------------------------------- mask @ V, added to out
// out[b, s=q, h*64+hd] += sum_k mask[q,k] * V[b,h,k,hd]   (V read from Vt)
__global__ __launch_bounds__(128) void mask_v_gemm_add(
    const _Float16* __restrict__ maskH, // [S,S]
    const _Float16* __restrict__ Vt,    // [B,H,64,S]
    float* __restrict__ out)            // [B,S,1024]
{
  const int lane = threadIdx.x & 31;
  const int wid  = blockIdx.x * (blockDim.x >> 5) + (threadIdx.x >> 5);
  const int qt = wid & (SEQ / 16 - 1);
  const int h  = (wid >> 6) & (N_HEAD - 1);
  const int b  = wid >> 10;
  const _Float16* Abase = maskH + (size_t)qt * 16 * SEQ;
  const _Float16* Vbase = Vt + ((size_t)b * N_HEAD + h) * HEAD_DIM * SEQ;

  v8f c0 = {}, c1 = {}, c2 = {}, c3 = {};
  for (int kt = 0; kt < SEQ / 32; ++kt) {
    v16h a = load_a_frag(Abase + kt * 32, SEQ, lane);
    c0 = WMMA_F16(a, load_b_frag(Vbase + kt * 32,           SEQ, lane), c0);
    c1 = WMMA_F16(a, load_b_frag(Vbase + 16*SEQ + kt * 32,  SEQ, lane), c1);
    c2 = WMMA_F16(a, load_b_frag(Vbase + 32*SEQ + kt * 32,  SEQ, lane), c2);
    c3 = WMMA_F16(a, load_b_frag(Vbase + 48*SEQ + kt * 32,  SEQ, lane), c3);
  }

  const int hd   = lane & 15;
  const int half = lane >> 4;
  #pragma unroll
  for (int r = 0; r < 8; ++r) {
    float* row = out + ((size_t)b * SEQ + qt * 16 + r + 8 * half) * D_MODEL
                     + h * HEAD_DIM + hd;
    row[0]  += c0[r];
    row[16] += c1[r];
    row[32] += c2[r];
    row[48] += c3[r];
  }
}

// ---------------------------------------------------------------- launch
extern "C" void kernel_launch(void* const* d_in, const int* in_sizes, int n_in,
                              void* d_out, int out_size, void* d_ws, size_t ws_size,
                              hipStream_t stream) {
  const float* x    = (const float*)d_in[0];
  const float* y    = (const float*)d_in[1];
  const float* z    = (const float*)d_in[2];
  const float* mask = (const float*)d_in[3];
  const float* wq_w = (const float*)d_in[4];
  const float* wq_b = (const float*)d_in[5];
  const float* wk_w = (const float*)d_in[6];
  const float* wk_b = (const float*)d_in[7];
  const float* wv_w = (const float*)d_in[8];
  const float* wv_b = (const float*)d_in[9];
  float* out = (float*)d_out;

  const size_t ACT = (size_t)BATCH * SEQ * D_MODEL;   // 4M elems
  const size_t WEI = (size_t)D_MODEL * D_MODEL;       // 1M elems
  const size_t MSK = (size_t)SEQ * SEQ;               // 1M elems

  _Float16* p   = (_Float16*)d_ws;
  _Float16* Xh  = p; p += ACT;
  _Float16* Yh  = p; p += ACT;
  _Float16* Zh  = p; p += ACT;
  _Float16* Wqh = p; p += WEI;
  _Float16* Wkh = p; p += WEI;
  _Float16* Wvh = p; p += WEI;
  _Float16* Mh  = p; p += MSK;
  _Float16* Qh  = p; p += ACT;
  _Float16* Kh  = p; p += ACT;
  _Float16* Vt  = p; p += ACT;   // V transposed: [B,H,64,S]

  cvt_f32_to_f16<<<4096, 256, 0, stream>>>(x,    Xh,  (int)ACT);
  cvt_f32_to_f16<<<4096, 256, 0, stream>>>(y,    Yh,  (int)ACT);
  cvt_f32_to_f16<<<4096, 256, 0, stream>>>(z,    Zh,  (int)ACT);
  cvt_f32_to_f16<<<1024, 256, 0, stream>>>(wq_w, Wqh, (int)WEI);
  cvt_f32_to_f16<<<1024, 256, 0, stream>>>(wk_w, Wkh, (int)WEI);
  cvt_f32_to_f16<<<1024, 256, 0, stream>>>(wv_w, Wvh, (int)WEI);
  cvt_f32_to_f16<<<1024, 256, 0, stream>>>(mask, Mh,  (int)MSK);

  // 2048 (32x64)-tile waves per projection, 4 waves/block
  proj_gemm_f16<<<512, 128, 0, stream>>>(Xh, Wqh, wq_b, Qh, 0);
  proj_gemm_f16<<<512, 128, 0, stream>>>(Yh, Wkh, wk_b, Kh, 0);
  proj_gemm_f16<<<512, 128, 0, stream>>>(Zh, Wvh, wv_b, Vt, 1);

  // 1024 blocks x 4 waves: one (b,h, 4-q-tile group) per block
  flash_attn_f16 <<<1024, 128, 0, stream>>>(Qh, Kh, Vt, out);
  mask_v_gemm_add<<<1024, 128, 0, stream>>>(Mh, Vt, out);
}
